// LSTM_AE_6614249635851
// MI455X (gfx1250) — matrix operand order repaired and checked by reading, not compiled, not run on hardware
//
#include <hip/hip_runtime.h>
#include <hip/hip_bf16.h>
#include <stdint.h>
#include <stddef.h>

// ---------------------------------------------------------------------------
// LSTM autoencoder for MI455X (gfx1250, wave32, WMMA).
// B=256, T=512, D=128, H=256.
//
// 16 persistent workgroups, each owns 16 batch rows and runs the full
// encoder+decoder time loop with h/c state in LDS. Weights converted once to
// bf16 (~3.7MB, L2-resident) and re-streamed every step into
// v_wmma_f32_16x16x32_bf16 by all workgroups (shared L2 lines). Each of 16
// waves owns a 16-wide hidden slice and computes i,f,g,o for that slice, so
// the LSTM nonlinearity stays in WMMA accumulator registers.
//
// Codegen notes (from rounds 1-2 disasm):
//  * An opaque per-iteration integer offset (asm "+s") stops LICM from
//    hoisting weight fragments across the t-loop (round-1 scratch spills),
//    while keeping the pointer a pure GEP off the kernarg so the loads stay
//    GLOBAL (round-2 laundered pointers degraded to FLAT, which ties up
//    DScnt as well as LOADcnt).
//  * Accumulators are zero-initialized so the first WMMA per gate folds
//    SRC2 to the inline 0 constant; biases are added in the register
//    epilogue instead of being splatted into 8 VGPRs every step.
// ---------------------------------------------------------------------------

#define Bb 256
#define Tt 512
#define Dd 128
#define Hh 256

#define MT        16    // batch rows per workgroup
#define NTHREADS  512   // 16 waves of 32
#define HPAD      264   // bf16 elems per h row (pad 256 -> bank-conflict free)
#define CPAD      260   // f32 elems per c row
#define XPAD      136   // bf16 elems per x/y row (pad 128)

typedef __attribute__((ext_vector_type(16))) __bf16 v16bf;
typedef __attribute__((ext_vector_type(8)))  __bf16 v8bf;
typedef __attribute__((ext_vector_type(8)))  float  v8f;

__device__ __forceinline__ float sigf(float x) {
  return 1.0f / (1.0f + __expf(-x));
}
// tanh(x) = 1 - 2/(exp(2x)+1): monotone-safe at +-inf, uses v_exp_f32.
__device__ __forceinline__ float tanh_(float x) {
  return 1.0f - 2.0f / (__expf(2.0f * x) + 1.0f);
}

__device__ __forceinline__ v8f wmma_bf16(v16bf a, v16bf b, v8f c) {
  // (neg_a, A, neg_b, B, c_mod, C, reuse_a, reuse_b)
  return __builtin_amdgcn_wmma_f32_16x16x32_bf16(false, a, false, b,
                                                 (short)0, c, false, false);
}

// Opaque zero, redefined every call: derived addresses are not loop-invariant
// (defeats LICM) but remain plain GEPs off the kernarg (stay addrspace(1)).
__device__ __forceinline__ size_t opaque_zero() {
  size_t z = 0;
  asm volatile("" : "+s"(z));
  return z;
}

// A fragment (16x32, MxK) from row-major [16][stride] bf16 (LDS):
//   lanes 0-15 : row M=lane,    K = k + {0..7, 16..23}
//   lanes 16-31: row M=lane&15, K = k + {8..15, 24..31}
__device__ __forceinline__ v16bf load_a(const __bf16* base, int stride,
                                        int lane, int k) {
  const int m    = lane & 15;
  const int koff = k + ((lane >> 4) << 3);
  const __bf16* p = base + m * stride + koff;
  union { v16bf v; v8bf h[2]; } u;
  u.h[0] = *reinterpret_cast<const v8bf*>(p);       // K koff..koff+7
  u.h[1] = *reinterpret_cast<const v8bf*>(p + 16);  // K koff+16..koff+23
  return u.v;
}

// B fragment (32x16, KxN) where B = W^T and W is row-major [N][Kdim] bf16:
// column n of B == row n of W (contiguous in K).
//   lanes 0-15 : col N=n0+lane,    K = k + {0..7, 16..23}
//   lanes 16-31: col N=n0+lane&15, K = k + {8..15, 24..31}
__device__ __forceinline__ v16bf load_b(const __bf16* W, int Kdim, int n0,
                                        int lane, int k) {
  const int n    = n0 + (lane & 15);
  const int koff = k + ((lane >> 4) << 3);
  const __bf16* p = W + n * Kdim + koff;
  union { v16bf v; v8bf h[2]; } u;
  u.h[0] = *reinterpret_cast<const v8bf*>(p);
  u.h[1] = *reinterpret_cast<const v8bf*>(p + 16);
  return u.v;
}

// One LSTM cell for this workgroup's 16 batch rows.
// gates = a1 @ Wih^T + hprev @ Whh^T + bih + bhh, then pointwise update.
// Wave w owns hidden cols [w*16, w*16+16): acc[g] is gate g for that slice,
// so i/f/g/o align per (reg, lane) and the update is register-resident.
template <int K1>
__device__ __forceinline__ void lstm_cell(
    const __bf16* a1, int a1_stride,
    const __bf16* Wih,
    const __bf16* hprev,
    const __bf16* Whh,
    const float* __restrict__ bih, const float* __restrict__ bhh,
    __bf16* hout, float* cbuf, int lane, int wave)
{
  const int lm = lane & 15;
  const int nb = wave << 4;  // this wave's 16-wide hidden slice
  float bb[4];
  v8f acc[4];
  const v8f vzero = {0.f, 0.f, 0.f, 0.f, 0.f, 0.f, 0.f, 0.f};
#pragma unroll
  for (int g = 0; g < 4; ++g) {
    const int n = (g << 8) + nb + lm;  // gate g, col in [0,1024)
    bb[g] = bih[n] + bhh[n];
    acc[g] = vzero;  // first WMMA folds SRC2 to inline 0
  }
  // gates += a1 @ Wih^T  (K = K1)
#pragma unroll
  for (int k = 0; k < K1; k += 32) {
    const v16bf a = load_a(a1, a1_stride, lane, k);
    v16bf bfr[4];
#pragma unroll
    for (int g = 0; g < 4; ++g)
      bfr[g] = load_b(Wih, K1, (g << 8) + nb, lane, k);
#pragma unroll
    for (int g = 0; g < 4; ++g)
      acc[g] = wmma_bf16(a, bfr[g], acc[g]);
  }
  // gates += hprev @ Whh^T  (K = 256)
#pragma unroll
  for (int k = 0; k < Hh; k += 32) {
    const v16bf a = load_a(hprev, HPAD, lane, k);
    v16bf bfr[4];
#pragma unroll
    for (int g = 0; g < 4; ++g)
      bfr[g] = load_b(Whh, Hh, (g << 8) + nb, lane, k);
#pragma unroll
    for (int g = 0; g < 4; ++g)
      acc[g] = wmma_bf16(a, bfr[g], acc[g]);
  }
  __syncthreads();  // all reads of hprev done before hout overwrite

  // Pointwise LSTM update on accumulator registers (bias folded in here).
  // C/D layout: reg r, lane l -> (M = r + 8*(l>=16), N = n_tile + (l&15)).
  const int rowoff = (lane >> 4) << 3;
  const int j      = nb + lm;  // hidden index owned by this lane
#pragma unroll
  for (int r = 0; r < 8; ++r) {
    const int m  = r + rowoff;
    const float gi = sigf(acc[0][r] + bb[0]);
    const float gf = sigf(acc[1][r] + bb[1]);
    const float gg = tanh_(acc[2][r] + bb[2]);
    const float go = sigf(acc[3][r] + bb[3]);
    const float cn = gf * cbuf[m * CPAD + j] + gi * gg;
    cbuf[m * CPAD + j] = cn;
    hout[m * HPAD + j] = (__bf16)(go * tanh_(cn));
  }
  __syncthreads();  // new h visible to all waves
}

// y = h2 @ fc_W^T + fc_b ; write f32 to global out, bf16 feedback into xin.
__device__ __forceinline__ void fc_head(
    const __bf16* h2, const __bf16* fcW,
    const float* __restrict__ fcb,
    float* __restrict__ out, __bf16* xin, int b0, int t, int lane, int wave)
{
  if (wave < 8) {  // 8 waves cover 128 output cols; branch is wave-uniform
    const int lm = lane & 15;
    const int n0 = wave << 4;
    const float bb = fcb[n0 + lm];
    v8f acc = {0.f, 0.f, 0.f, 0.f, 0.f, 0.f, 0.f, 0.f};
#pragma unroll
    for (int k = 0; k < Hh; k += 32)
      acc = wmma_bf16(load_a(h2, HPAD, lane, k),
                      load_b(fcW, Hh, n0, lane, k), acc);
    const int rowoff = (lane >> 4) << 3;
#pragma unroll
    for (int r = 0; r < 8; ++r) {
      const int m = r + rowoff;
      const float y = acc[r] + bb;
      out[((size_t)(b0 + m) * Tt + t) * Dd + n0 + lm] = y;
      xin[m * XPAD + n0 + lm] = (__bf16)y;
    }
  }
  __syncthreads();  // y feedback visible before next step's layer-0 GEMM
}

__global__ void __launch_bounds__(NTHREADS, 1) lstm_ae_main(
    const float* __restrict__ x,
    const __bf16* __restrict__ wts,
    const float* __restrict__ e_bih0, const float* __restrict__ e_bhh0,
    const float* __restrict__ e_bih1, const float* __restrict__ e_bhh1,
    const float* __restrict__ d_bih0, const float* __restrict__ d_bhh0,
    const float* __restrict__ d_bih1, const float* __restrict__ d_bhh1,
    const float* __restrict__ fc_b,
    float* __restrict__ out)
{
  __shared__ __bf16 sh1[MT * HPAD];
  __shared__ __bf16 sh2[MT * HPAD];
  __shared__ __bf16 sxin[MT * XPAD];
  __shared__ float  sc1[MT * CPAD];
  __shared__ float  sc2[MT * CPAD];

  const int tid  = threadIdx.x;
  const int lane = tid & 31;
  const int wave = tid >> 5;
  const int b0   = blockIdx.x * MT;

  for (int i = tid; i < MT * HPAD; i += NTHREADS) {
    sh1[i] = (__bf16)0.0f; sh2[i] = (__bf16)0.0f;
  }
  for (int i = tid; i < MT * CPAD; i += NTHREADS) {
    sc1[i] = 0.0f; sc2[i] = 0.0f;
  }
  __syncthreads();

  // ---------------- encoder ----------------
  const int xm = tid >> 5;         // 0..15 (row)
  const int xc = (tid & 31) << 2;  // 0..124 (col base, 4 per thread)
  for (int t = 0; t < Tt; ++t) {
    // Opaque offset: weight loads stay inside the t-loop (global_load from
    // L2-resident weights) instead of being hoisted and spilled to scratch.
    const __bf16* wb = wts + opaque_zero();
    const __bf16* eWih0 = wb + 0;        // [1024][128]
    const __bf16* eWhh0 = wb + 131072;   // [1024][256]
    const __bf16* eWih1 = wb + 393216;   // [1024][256]
    const __bf16* eWhh1 = wb + 655360;   // [1024][256]

    const float* xp = x + ((size_t)(b0 + xm) * Tt + t) * Dd + xc;
#pragma unroll
    for (int q = 0; q < 4; ++q)
      sxin[xm * XPAD + xc + q] = (__bf16)xp[q];
    __syncthreads();
    lstm_cell<Dd>(sxin, XPAD, eWih0, sh1, eWhh0, e_bih0, e_bhh0, sh1, sc1,
                  lane, wave);
    lstm_cell<Hh>(sh1, HPAD, eWih1, sh2, eWhh1, e_bih1, e_bhh1, sh2, sc2,
                  lane, wave);
  }

  // ---------------- decoder (autoregressive) ----------------
  for (int i = tid; i < MT * XPAD; i += NTHREADS) sxin[i] = (__bf16)0.0f;
  __syncthreads();
  for (int t = 0; t < Tt; ++t) {
    const __bf16* wb = wts + opaque_zero();
    const __bf16* dWih0 = wb + 917504;   // [1024][128]
    const __bf16* dWhh0 = wb + 1048576;  // [1024][256]
    const __bf16* dWih1 = wb + 1310720;  // [1024][256]
    const __bf16* dWhh1 = wb + 1572864;  // [1024][256]
    const __bf16* fcW   = wb + 1835008;  // [128][256]

    lstm_cell<Dd>(sxin, XPAD, dWih0, sh1, dWhh0, d_bih0, d_bhh0, sh1, sc1,
                  lane, wave);
    lstm_cell<Hh>(sh1, HPAD, dWih1, sh2, dWhh1, d_bih1, d_bhh1, sh2, sc2,
                  lane, wave);
    fc_head(sh2, fcW, fc_b, out, sxin, b0, t, lane, wave);
  }
}

// One-shot f32 -> bf16 weight conversion into workspace (L2-resident).
__global__ void cvt_f32_bf16(const float* __restrict__ src,
                             __bf16* __restrict__ dst, int n) {
  int i = blockIdx.x * blockDim.x + threadIdx.x;
  if (i < n) dst[i] = (__bf16)src[i];
}

extern "C" void kernel_launch(void* const* d_in, const int* in_sizes, int n_in,
                              void* d_out, int out_size, void* d_ws,
                              size_t ws_size, hipStream_t stream) {
  (void)in_sizes; (void)n_in; (void)out_size; (void)ws_size;
  __bf16* wsb = (__bf16*)d_ws;

  // input order: x, e_{Wih0,Whh0,bih0,bhh0}, e_{..1}, d_{..0}, d_{..1}, fc_W, fc_b
  static const int    widx[9] = {1, 2, 5, 6, 9, 10, 13, 14, 17};
  static const size_t woff[9] = {0, 131072, 393216, 655360, 917504,
                                 1048576, 1310720, 1572864, 1835008};
  static const int    wn[9]   = {131072, 262144, 262144, 262144, 131072,
                                 262144, 262144, 262144, 32768};
  for (int i = 0; i < 9; ++i) {
    const int n = wn[i];
    cvt_f32_bf16<<<(n + 255) / 256, 256, 0, stream>>>(
        (const float*)d_in[widx[i]], wsb + woff[i], n);
  }

  lstm_ae_main<<<Bb / MT, NTHREADS, 0, stream>>>(
      (const float*)d_in[0], wsb,
      (const float*)d_in[3],  (const float*)d_in[4],   // e_bih0, e_bhh0
      (const float*)d_in[7],  (const float*)d_in[8],   // e_bih1, e_bhh1
      (const float*)d_in[11], (const float*)d_in[12],  // d_bih0, d_bhh0
      (const float*)d_in[15], (const float*)d_in[16],  // d_bih1, d_bhh1
      (const float*)d_in[18],                          // fc_b
      (float*)d_out);
}
